// Encoder_79130477461967
// MI455X (gfx1250) — matrix operand order repaired and checked by reading, not compiled, not run on hardware
//
#include <hip/hip_runtime.h>

// ---------------------------------------------------------------------------
// Types for CDNA5 WMMA (wave32): bf16 A/B, f32 accumulator
// ---------------------------------------------------------------------------
typedef __attribute__((ext_vector_type(16))) __bf16 bf16x16;
typedef __attribute__((ext_vector_type(8)))  float  f32x8;

#define B_SZ    4
#define N_SEQ   128
#define IN_D    64
#define C_D     128      // WC_C / LP_C
#define BC_D    512
#define LP_IN   (BC_D + C_D)   // 640
#define KPAD    9
#define T_STEPS 127

#define WMMA_BF16(acc, a, b) \
  __builtin_amdgcn_wmma_f32_16x16x32_bf16(false, (a), false, (b), (short)0, (acc), false, false)

// --- A-matrix (16x32, bf16) lane layout: lane m = L%16, half h = L/16.
// element e (of 16 per lane): vgpr v=e/2 ; k = (v<4 ? 2v : 16+2(v-4)) + (e&1) + 8h
static __device__ __forceinline__ int akmap(int e, int h) {
  int v = e >> 1;
  return ((v < 4) ? (2 * v) : (16 + 2 * (v - 4))) + (e & 1) + h * 8;
}

static __device__ __forceinline__ bf16x16
ldA_bf16(const __bf16* src, int ld, int m0, int k0, int lane) {
  int m = m0 + (lane & 15), h = (lane >> 4) & 1;
  bf16x16 a;
#pragma unroll
  for (int e = 0; e < 16; ++e) a[e] = src[m * ld + k0 + akmap(e, h)];
  return a;
}

static __device__ __forceinline__ bf16x16
ldA_f32(const float* src, int ld, int m0, int k0, int lane) {
  int m = m0 + (lane & 15), h = (lane >> 4) & 1;
  bf16x16 a;
#pragma unroll
  for (int e = 0; e < 16; ++e) a[e] = (__bf16)src[m * ld + k0 + akmap(e, h)];
  return a;
}

// --- B-matrix fragment-major layout: fragment (kblk = k0/32, nblk = n0/16),
// lane = 16*h + n_local holds its 16 bf16 contiguously (32 bytes).
// Whole wave reads one contiguous 1KB burst -> 2x global_load_b128 per lane.
static __device__ __forceinline__ bf16x16
ldBf(const __bf16* Wf, int ntiles, int kblk, int nblk, int lane) {
  return *(const bf16x16*)(Wf + ((((size_t)kblk * ntiles + nblk) * 32 + lane) << 4));
}

// ---------------------------------------------------------------------------
// Kernel 1: weight f32 -> bf16 fragment-major swizzle + nt[] scan
// ---------------------------------------------------------------------------
static __device__ __forceinline__ void
convert_frag(const float* src, __bf16* dst, int K, int N, int gid, int str) {
  int total = K * N;
  int ntiles = N >> 4;
  for (int idx = gid; idx < total; idx += str) {
    int k = idx / N, n = idx - k * N;
    int kblk = k >> 5, kin = k & 31;
    int h = (kin >> 3) & 1, e = (kin & 7) + ((kin >> 4) << 3);
    int nblk = n >> 4, nl = n & 15;
    int lane = h * 16 + nl;
    dst[(((size_t)kblk * ntiles + nblk) * 32 + lane) * 16 + e] = (__bf16)src[idx];
  }
}

__global__ void prep_kernel(
    const float* wcW1, const float* wcW2, const float* usW1, const float* usW2,
    const float* bcW1, const float* bcW2, const float* lpW1, const float* lpW2,
    __bf16* wcW1f, __bf16* wcW2f, __bf16* usW1f, __bf16* usW2f,
    __bf16* bcW1f, __bf16* bcW2f, __bf16* lpW1f, __bf16* lpW2f,
    const int* y, int* nt) {
  int gid = blockIdx.x * blockDim.x + threadIdx.x;
  int str = gridDim.x * blockDim.x;
  convert_frag(wcW1, wcW1f, IN_D, C_D, gid, str);
  convert_frag(wcW2, wcW2f, C_D, C_D, gid, str);
  convert_frag(usW1, usW1f, IN_D, C_D, gid, str);
  convert_frag(usW2, usW2f, C_D, C_D, gid, str);
  convert_frag(bcW1, bcW1f, C_D, BC_D, gid, str);
  convert_frag(bcW2, bcW2f, BC_D, BC_D, gid, str);
  convert_frag(lpW1, lpW1f, LP_IN, C_D, gid, str);
  convert_frag(lpW2, lpW2f, C_D, C_D, gid, str);
  if (gid == 0) {
    int mx = 0;
    for (int t = 0; t < T_STEPS; ++t) { int v = y[t]; mx = v > mx ? v : mx; nt[t] = mx + 1; }
  }
}

// ---------------------------------------------------------------------------
// Kernel 2: 2-layer MLP (64->128 prelu ->128) over 512 rows, WMMA bf16
// Launched twice (wc params, us params). grid=8 blocks x 64 rows.
// ---------------------------------------------------------------------------
__global__ __launch_bounds__(256) void mlp2_kernel(
    const float* X, const __bf16* W1f, const float* b1, const float* a1p,
    const __bf16* W2f, const float* b2, float* out) {
  __shared__ float Xs[64 * IN_D];    // 16 KB
  __shared__ float H1[64 * C_D];     // 32 KB
  int tid = threadIdx.x;
  int r0 = blockIdx.x * 64;
  int wave = tid >> 5, lane = tid & 31;
  float a1 = a1p[0];

  for (int idx = tid; idx < 64 * IN_D; idx += 256) {
    int r = idx >> 6, c = idx & 63;
    Xs[idx] = X[(size_t)(r0 + r) * IN_D + c];
  }
  __syncthreads();

  // layer 1: [64x64] @ [64x128], bias + prelu -> H1 (f32)
  for (int tile = wave; tile < 4 * 8; tile += 8) {
    int mt = tile >> 3, ntl = tile & 7;
    f32x8 acc = {0.f, 0.f, 0.f, 0.f, 0.f, 0.f, 0.f, 0.f};
#pragma unroll
    for (int k0 = 0; k0 < IN_D; k0 += 32) {
      bf16x16 a = ldA_f32(Xs, IN_D, mt * 16, k0, lane);
      bf16x16 b = ldBf(W1f, 8, k0 >> 5, ntl, lane);
      acc = WMMA_BF16(acc, a, b);
    }
    int n = ntl * 16 + (lane & 15), h = lane >> 4;
#pragma unroll
    for (int v = 0; v < 8; ++v) {
      int m = mt * 16 + v + 8 * h;
      float x = acc[v] + b1[n];
      H1[m * C_D + n] = (x >= 0.f) ? x : a1 * x;
    }
  }
  __syncthreads();

  // layer 2: [64x128] @ [128x128] + bias -> out (f32, global)
  for (int tile = wave; tile < 4 * 8; tile += 8) {
    int mt = tile >> 3, ntl = tile & 7;
    f32x8 acc = {0.f, 0.f, 0.f, 0.f, 0.f, 0.f, 0.f, 0.f};
#pragma unroll
    for (int k0 = 0; k0 < C_D; k0 += 32) {
      bf16x16 a = ldA_f32(H1, C_D, mt * 16, k0, lane);
      bf16x16 b = ldBf(W2f, 8, k0 >> 5, ntl, lane);
      acc = WMMA_BF16(acc, a, b);
    }
    int n = ntl * 16 + (lane & 15), h = lane >> 4;
#pragma unroll
    for (int v = 0; v < 8; ++v) {
      int m = mt * 16 + v + 8 * h;
      out[(size_t)(r0 + m) * C_D + n] = acc[v] + b2[n];
    }
  }
}

// ---------------------------------------------------------------------------
// Kernel 3: scan. Thread = (b,c). Materialize WC_t [127][4][9][128] and
// US_t [127][4][128]. Teacher-forced -> pure prefix sums, no logits needed.
// ---------------------------------------------------------------------------
__global__ void scan_kernel(const float* wcf, const float* usf, const int* y,
                            float* WCs, float* USs) {
  int gid = blockIdx.x * blockDim.x + threadIdx.x;
  if (gid >= B_SZ * C_D) return;
  int b = gid >> 7, c = gid & 127;

  float slots[KPAD];
#pragma unroll
  for (int j = 0; j < KPAD; ++j) slots[j] = 0.f;
  slots[0] = wcf[(size_t)(b * N_SEQ + 0) * C_D + c];  // WC0[:,0] = wc[:,0]

  float u = 0.f;
  for (int n = 2; n < N_SEQ; ++n) u += usf[(size_t)(b * N_SEQ + n) * C_D + c];

  for (int t = 0; t < T_STEPS; ++t) {
    float w = wcf[(size_t)(b * N_SEQ + t) * C_D + c];
    int yt = y[t];
#pragma unroll
    for (int j = 0; j < KPAD; ++j) if (j == yt) slots[j] += w;
    u -= usf[(size_t)(b * N_SEQ + (t + 1)) * C_D + c];
    float uo = (t == T_STEPS - 1) ? 0.f : u;
#pragma unroll
    for (int j = 0; j < KPAD; ++j)
      WCs[((size_t)t * (B_SZ * KPAD) + b * KPAD + j) * C_D + c] = slots[j];
    USs[(size_t)t * (B_SZ * C_D) + b * C_D + c] = uo;
  }
}

// ---------------------------------------------------------------------------
// Kernel 4: one workgroup per step. BC MLP on 80 rows (20/batch:
// 8 base slots, 8 slots+wc_i, 1 wc_i, 3 pad), incremental-sum trick,
// LP head on 48 rows, log-softmax, per-step NLL partial.
// Dynamic LDS layout (bytes):
//   OUT2 f32 [80][512]  @ 0       (163840)   (reused: lpH2 f32 [48][128])
//   H1  bf16 [80][512]  @ 163840  ( 81920)   (reused: lpIn bf16 [48][640])
//   A1  bf16 [80][128]  @ 245760  ( 20480)   (reused: lpH1 bf16 [48][128])
//   Sb  f32  [4][512]   @ 266240  (  8192)
//   USl f32  [4][128]   @ 274432  (  2048)
//   lpO f32  [48]       @ 276480  (   256)
// ---------------------------------------------------------------------------
#define SMEM_STEP_BYTES 276736

__global__ __launch_bounds__(256) void step_kernel(
    const float* WCs, const float* USs, const int* ntg, const int* y,
    const float* wcf,
    const __bf16* bcW1f, const float* bcb1, const float* bca1,
    const __bf16* bcW2f, const float* bcb2,
    const __bf16* lpW1f, const float* lpb1, const float* lpa1,
    const __bf16* lpW2f, const float* lpb2, const float* lpa2,
    const float* lpW3, float* nll_steps) {
  extern __shared__ char smem[];
  float*  OUT2 = (float*)(smem);
  __bf16* H1   = (__bf16*)(smem + 163840);
  __bf16* A1   = (__bf16*)(smem + 245760);
  float*  Sb   = (float*)(smem + 266240);
  float*  USl  = (float*)(smem + 274432);
  float*  lpO  = (float*)(smem + 276480);

  int t = blockIdx.x, i = t + 1;
  int tid = threadIdx.x, wave = tid >> 5, lane = tid & 31;
  int ntc = ntg[t];
  int yi = y[i];

  // phase 0: build BC input rows (bf16) + stage US
  const float* WCt = WCs + (size_t)t * (B_SZ * KPAD * C_D);
  for (int idx = tid; idx < 80 * C_D; idx += 256) {
    int r = idx >> 7, c = idx & 127;
    int b = r / 20, lr = r % 20;
    float wci = wcf[(size_t)(b * N_SEQ + i) * C_D + c];
    float v;
    if (lr < 8)        v = WCt[(b * KPAD + lr) * C_D + c];
    else if (lr < 16)  v = WCt[(b * KPAD + (lr - 8)) * C_D + c] + wci;
    else if (lr == 16) v = wci;
    else               v = 0.f;
    A1[idx] = (__bf16)v;
  }
  for (int idx = tid; idx < B_SZ * C_D; idx += 256)
    USl[idx] = USs[(size_t)t * (B_SZ * C_D) + idx];
  __syncthreads();

  // phase 1: BC layer 1: [80x128]@[128x512] + bias, prelu -> H1 (bf16)
  float a1c = bca1[0];
  for (int tile = wave; tile < 5 * 32; tile += 8) {
    int mt = tile >> 5, ntl = tile & 31;
    f32x8 acc = {0.f, 0.f, 0.f, 0.f, 0.f, 0.f, 0.f, 0.f};
#pragma unroll
    for (int k0 = 0; k0 < C_D; k0 += 32) {
      bf16x16 a = ldA_bf16(A1, C_D, mt * 16, k0, lane);
      bf16x16 b = ldBf(bcW1f, 32, k0 >> 5, ntl, lane);
      acc = WMMA_BF16(acc, a, b);
    }
    int n = ntl * 16 + (lane & 15), h = lane >> 4;
#pragma unroll
    for (int v = 0; v < 8; ++v) {
      int m = mt * 16 + v + 8 * h;
      float x = acc[v] + bcb1[n];
      H1[m * BC_D + n] = (__bf16)((x >= 0.f) ? x : a1c * x);
    }
  }
  __syncthreads();

  // phase 2: BC layer 2: [80x512]@[512x512] + bias -> OUT2 (f32)
  for (int tile = wave; tile < 5 * 32; tile += 8) {
    int mt = tile >> 5, ntl = tile & 31;
    f32x8 acc = {0.f, 0.f, 0.f, 0.f, 0.f, 0.f, 0.f, 0.f};
#pragma unroll
    for (int k0 = 0; k0 < BC_D; k0 += 32) {
      bf16x16 a = ldA_bf16(H1, BC_D, mt * 16, k0, lane);
      bf16x16 b = ldBf(bcW2f, 32, k0 >> 5, ntl, lane);
      acc = WMMA_BF16(acc, a, b);
    }
    int n = ntl * 16 + (lane & 15), h = lane >> 4;
#pragma unroll
    for (int v = 0; v < 8; ++v) {
      int m = mt * 16 + v + 8 * h;
      OUT2[m * BC_D + n] = acc[v] + bcb2[n];
    }
  }
  __syncthreads();

  // phase 3: S_base[b] = sum_{j<ntc} BC(WC[b][j])
  for (int idx = tid; idx < B_SZ * BC_D; idx += 256) {
    int b = idx >> 9, c = idx & 511;
    float s = 0.f;
    for (int j = 0; j < ntc; ++j) s += OUT2[(b * 20 + j) * BC_D + c];
    Sb[idx] = s;
  }
  __syncthreads();

  // phase 4: LP input rows [48][640] bf16 (12 candidate rows per batch)
  __bf16* lpIn = H1;  // reuse
  for (int idx = tid; idx < 48 * LP_IN; idx += 256) {
    int r = idx / LP_IN, c = idx % LP_IN;
    int b = r / 12, cand = r % 12;
    float v = 0.f;
    if (c < BC_D) {
      if (cand < ntc)
        v = Sb[b * BC_D + c] - OUT2[(b * 20 + cand) * BC_D + c]
                             + OUT2[(b * 20 + 8 + cand) * BC_D + c];
      else if (cand == ntc)
        v = Sb[b * BC_D + c] + OUT2[(b * 20 + 16) * BC_D + c];
    } else {
      v = (cand <= ntc) ? USl[b * C_D + (c - BC_D)] : 0.f;
    }
    lpIn[idx] = (__bf16)v;
  }
  __syncthreads();

  // phase 5: LP layer 1: [48x640]@[640x128] + bias, prelu -> lpH1 (bf16)
  __bf16* lpH1 = A1;  // reuse
  float la1 = lpa1[0];
  for (int tile = wave; tile < 3 * 8; tile += 8) {
    int mt = tile >> 3, ntl = tile & 7;
    f32x8 acc = {0.f, 0.f, 0.f, 0.f, 0.f, 0.f, 0.f, 0.f};
    for (int k0 = 0; k0 < LP_IN; k0 += 32) {
      bf16x16 a = ldA_bf16(lpIn, LP_IN, mt * 16, k0, lane);
      bf16x16 b = ldBf(lpW1f, 8, k0 >> 5, ntl, lane);
      acc = WMMA_BF16(acc, a, b);
    }
    int n = ntl * 16 + (lane & 15), h = lane >> 4;
#pragma unroll
    for (int v = 0; v < 8; ++v) {
      int m = mt * 16 + v + 8 * h;
      float x = acc[v] + lpb1[n];
      lpH1[m * C_D + n] = (__bf16)((x >= 0.f) ? x : la1 * x);
    }
  }
  __syncthreads();

  // phase 6: LP layer 2: [48x128]@[128x128] + bias, prelu -> lpH2 (f32)
  float* lpH2 = OUT2;  // reuse
  float la2 = lpa2[0];
  for (int tile = wave; tile < 3 * 8; tile += 8) {
    int mt = tile >> 3, ntl = tile & 7;
    f32x8 acc = {0.f, 0.f, 0.f, 0.f, 0.f, 0.f, 0.f, 0.f};
#pragma unroll
    for (int k0 = 0; k0 < C_D; k0 += 32) {
      bf16x16 a = ldA_bf16(lpH1, C_D, mt * 16, k0, lane);
      bf16x16 b = ldBf(lpW2f, 8, k0 >> 5, ntl, lane);
      acc = WMMA_BF16(acc, a, b);
    }
    int n = ntl * 16 + (lane & 15), h = lane >> 4;
#pragma unroll
    for (int v = 0; v < 8; ++v) {
      int m = mt * 16 + v + 8 * h;
      float x = acc[v] + lpb2[n];
      lpH2[m * C_D + n] = (x >= 0.f) ? x : la2 * x;
    }
  }
  __syncthreads();

  // phase 7: LP layer 3 (128 -> 1, no bias)
  if (tid < 48) {
    float s = 0.f;
    for (int c = 0; c < C_D; ++c) s += lpH2[tid * C_D + c] * lpW3[c];
    lpO[tid] = s;
  }
  __syncthreads();

  // phase 8: log-softmax over ntc+1 candidates, per-step NLL partial
  if (tid == 0) {
    float tot = 0.f;
    for (int b = 0; b < B_SZ; ++b) {
      float mx = -3.0e38f;
      for (int cand = 0; cand <= ntc; ++cand) {
        float v = lpO[b * 12 + cand];
        mx = v > mx ? v : mx;
      }
      float se = 0.f;
      for (int cand = 0; cand <= ntc; ++cand) se += __expf(lpO[b * 12 + cand] - mx);
      float lse = mx + __logf(se);
      tot += lse - lpO[b * 12 + yi];
    }
    nll_steps[t] = tot * (1.0f / B_SZ);
  }
}

// ---------------------------------------------------------------------------
// Kernel 5: deterministic reduction of 127 per-step partials
// ---------------------------------------------------------------------------
__global__ void reduce_kernel(const float* nll_steps, float* out) {
  __shared__ float sh[128];
  int tid = threadIdx.x;
  sh[tid] = (tid < T_STEPS) ? nll_steps[tid] : 0.f;
  __syncthreads();
  for (int o = 64; o > 0; o >>= 1) {
    if (tid < o) sh[tid] += sh[tid + o];
    __syncthreads();
  }
  if (tid == 0) out[0] = sh[0];
}

// ---------------------------------------------------------------------------
// Host launcher
// ---------------------------------------------------------------------------
extern "C" void kernel_launch(void* const* d_in, const int* in_sizes, int n_in,
                              void* d_out, int out_size, void* d_ws, size_t ws_size,
                              hipStream_t stream) {
  (void)in_sizes; (void)n_in; (void)out_size; (void)ws_size;
  const float* X    = (const float*)d_in[0];
  const int*   y    = (const int*)d_in[1];
  const float* wcW1 = (const float*)d_in[2];
  const float* wcb1 = (const float*)d_in[3];
  const float* wca1 = (const float*)d_in[4];
  const float* wcW2 = (const float*)d_in[5];
  const float* wcb2 = (const float*)d_in[6];
  const float* usW1 = (const float*)d_in[7];
  const float* usb1 = (const float*)d_in[8];
  const float* usa1 = (const float*)d_in[9];
  const float* usW2 = (const float*)d_in[10];
  const float* usb2 = (const float*)d_in[11];
  const float* bcW1 = (const float*)d_in[12];
  const float* bcb1 = (const float*)d_in[13];
  const float* bca1 = (const float*)d_in[14];
  const float* bcW2 = (const float*)d_in[15];
  const float* bcb2 = (const float*)d_in[16];
  const float* lpW1 = (const float*)d_in[17];
  const float* lpb1 = (const float*)d_in[18];
  const float* lpa1 = (const float*)d_in[19];
  const float* lpW2 = (const float*)d_in[20];
  const float* lpb2 = (const float*)d_in[21];
  const float* lpa2 = (const float*)d_in[22];
  const float* lpW3 = (const float*)d_in[23];

  char* w = (char*)d_ws;
  size_t off = 0;
  auto carve = [&](size_t bytes) -> char* {
    char* p = w + off;
    off = (off + bytes + 255) & ~(size_t)255;
    return p;
  };
  __bf16* wcW1f = (__bf16*)carve(IN_D * C_D * 2);
  __bf16* wcW2f = (__bf16*)carve(C_D * C_D * 2);
  __bf16* usW1f = (__bf16*)carve(IN_D * C_D * 2);
  __bf16* usW2f = (__bf16*)carve(C_D * C_D * 2);
  __bf16* bcW1f = (__bf16*)carve(C_D * BC_D * 2);
  __bf16* bcW2f = (__bf16*)carve((size_t)BC_D * BC_D * 2);
  __bf16* lpW1f = (__bf16*)carve(LP_IN * C_D * 2);
  __bf16* lpW2f = (__bf16*)carve(C_D * C_D * 2);
  float*  wc_f  = (float*)carve((size_t)B_SZ * N_SEQ * C_D * 4);
  float*  us_f  = (float*)carve((size_t)B_SZ * N_SEQ * C_D * 4);
  float*  WCs   = (float*)carve((size_t)T_STEPS * B_SZ * KPAD * C_D * 4);
  float*  USs   = (float*)carve((size_t)T_STEPS * B_SZ * C_D * 4);
  int*    nt    = (int*)carve(T_STEPS * 4);
  float*  nlls  = (float*)carve(T_STEPS * 4);

  prep_kernel<<<64, 256, 0, stream>>>(
      wcW1, wcW2, usW1, usW2, bcW1, bcW2, lpW1, lpW2,
      wcW1f, wcW2f, usW1f, usW2f, bcW1f, bcW2f, lpW1f, lpW2f, y, nt);

  mlp2_kernel<<<8, 256, 0, stream>>>(X, wcW1f, wcb1, wca1, wcW2f, wcb2, wc_f);
  mlp2_kernel<<<8, 256, 0, stream>>>(X, usW1f, usb1, usa1, usW2f, usb2, us_f);

  scan_kernel<<<2, 256, 0, stream>>>(wc_f, us_f, y, WCs, USs);

  step_kernel<<<T_STEPS, 256, SMEM_STEP_BYTES, stream>>>(
      WCs, USs, nt, y, wc_f,
      bcW1f, bcb1, bca1, bcW2f, bcb2,
      lpW1f, lpb1, lpa1, lpW2f, lpb2, lpa2, lpW3, nlls);

  reduce_kernel<<<1, 128, 0, stream>>>(nlls, (float*)d_out);
}
